// BidirectionalMambaBlock_88725434401099
// MI455X (gfx1250) — matrix operand order repaired and checked by reading, not compiled, not run on hardware
//
#include <hip/hip_runtime.h>
#include <hip/hip_bf16.h>

// ---------------------------------------------------------------------------
// Bidirectional Mamba block for MI455X (gfx1250, wave32, WMMA).
// GEMMs: v_wmma_f32_16x16x32_f16, f32 accumulate, fused epilogues.
//        A tile staged to LDS via global_load_async_to_lds_b128 (double
//        buffered, s_wait_asynccnt) -> kills 4x redundant A traffic across
//        the N-direction waves; B read direct (weights L2-resident).
// Scan:  register-resident 16-wide SSM state per (b,d) thread, LDS B/C.
// ---------------------------------------------------------------------------

typedef __attribute__((ext_vector_type(16))) _Float16 v16h;
typedef __attribute__((ext_vector_type(8)))  _Float16 v8h;
typedef __attribute__((ext_vector_type(8)))  float    v8f;

#define B_   8
#define L_   1024
#define D_   512
#define DI_  2048
#define DS_  16
#define DTR_ 32
#define DFF_ 2048

#define AS3 __attribute__((address_space(3)))

__device__ __forceinline__ unsigned lds_addr32(const void* p) {
  // generic -> LDS address space, then to the 32-bit LDS byte offset the
  // async-load VDST operand expects.
  return (unsigned)(uintptr_t)(AS3 const char*)p;
}

__device__ __forceinline__ v16h cat8(v8h lo, v8h hi) {
  return __builtin_shufflevector(lo, hi, 0,1,2,3,4,5,6,7,8,9,10,11,12,13,14,15);
}

// A fragment (16x32 MxK) from an LDS tile row-major with 80B row stride:
// p already includes the (lane>>4)*16 byte K-chunk offset; chunks at +0,+32.
__device__ __forceinline__ v16h frag_a_lds(const char* p) {
  v8h lo = *reinterpret_cast<const v8h*>(p);
  v8h hi = *reinterpret_cast<const v8h*>(p + 32);
  return cat8(lo, hi);
}

// B fragment (32x16 KxN) from weights stored row-major (N,K):
// lane L<16 -> col=L, K=[0..7,16..23]; lane L>=16 -> col=L-16, K=[8..15,24..31].
__device__ __forceinline__ v16h load_frag_b(const _Float16* __restrict__ W,
                                            int ldb, int col0, int k0, int lane) {
  const int col = col0 + (lane & 15);
  const int kb  = k0 + ((lane >> 4) << 3);
  const _Float16* p = W + (long long)col * ldb + kb;
  v8h lo = *reinterpret_cast<const v8h*>(p);
  v8h hi = *reinterpret_cast<const v8h*>(p + 16);
  return cat8(lo, hi);
}

// ---------------- GEMM: C[M,N] = A[M,K] x W[N,K]^T, wave tile 32x64 --------
// EPI: 0 = f32 store            1 = f32 + f16 store
//      2 = softplus(v+bias)     3 = resid + v
//      4 = relu(v+bias) -> f16  5 = v + bias
template<int WAVES_M, int WAVES_N, int EPI>
__global__ __launch_bounds__(32 * WAVES_M * WAVES_N)
void gemm_wmma(const _Float16* __restrict__ A, int lda,
               const _Float16* __restrict__ W,
               float* __restrict__ Cout, _Float16* __restrict__ Ch,
               const float* __restrict__ bias, const float* __restrict__ resid,
               int M, int N, int K) {
  constexpr int NTHREADS  = 32 * WAVES_M * WAVES_N;
  constexpr int ROWS      = WAVES_M * 32;       // block-tile rows
  constexpr int ROWSTRIDE = 80;                 // 64B data + 16B pad (bank-clean)
  constexpr int BUFBYTES  = ROWS * ROWSTRIDE;
  constexpr int CPT       = (ROWS * 4) / NTHREADS;  // 16B chunks per thread
  static_assert(CPT >= 1, "staging needs >=1 chunk per thread");

  __shared__ __attribute__((aligned(16))) char ldsA[2 * BUFBYTES];

  const int lane = threadIdx.x & 31;
  const int wave = threadIdx.x >> 5;
  const int wm   = wave % WAVES_M;
  const int wn   = wave / WAVES_M;
  const int row0_blk = blockIdx.x * ROWS;
  const int row0     = row0_blk + wm * 32;
  const int col0     = (blockIdx.y * WAVES_N + wn) * 64;

  // ---- async stage of one 32-wide K slice of the block's A rows into LDS
  const unsigned lds_base = lds_addr32(&ldsA[0]);
  auto stage = [&](int buf, int k0) {
#pragma unroll
    for (int c = 0; c < CPT; ++c) {
      const int chunk = (int)threadIdx.x + c * NTHREADS;
      const int r  = chunk >> 2;        // row within block tile
      const int kc = chunk & 3;         // which 16B chunk of the 64B row
      const unsigned dst = lds_base + (unsigned)(buf * BUFBYTES + r * ROWSTRIDE + kc * 16);
      const unsigned long long src =
          (unsigned long long)(A + (long long)(row0_blk + r) * lda + k0 + kc * 8);
      asm volatile("global_load_async_to_lds_b128 %0, %1, off"
                   :: "v"(dst), "v"(src) : "memory");
    }
  };

  v8f acc[2][4];
#pragma unroll
  for (int i = 0; i < 2; ++i)
#pragma unroll
    for (int j = 0; j < 4; ++j)
#pragma unroll
      for (int e = 0; e < 8; ++e) acc[i][j][e] = 0.0f;

  stage(0, 0);
  asm volatile("s_wait_asynccnt 0" ::: "memory");
  __syncthreads();

  const int arow = wm * 32 + (lane & 15);
  const int akof = (lane >> 4) << 4;    // K-chunk byte offset within row

  int cur = 0;
  for (int k0 = 0; k0 < K; k0 += 32) {
    const bool more = (k0 + 32) < K;
    if (more) {
      stage(1 - cur, k0 + 32);  // async fill of the other buffer
      __builtin_prefetch(W + (long long)(col0 + (lane & 15)) * K + k0 + 32, 0, 1);
    }

    const char* pa = ldsA + cur * BUFBYTES + akof;
    v16h a0 = frag_a_lds(pa + (long long)arow * ROWSTRIDE);
    v16h a1 = frag_a_lds(pa + (long long)(arow + 16) * ROWSTRIDE);
    v16h b0 = load_frag_b(W, K, col0,      k0, lane);
    v16h b1 = load_frag_b(W, K, col0 + 16, k0, lane);
    v16h b2 = load_frag_b(W, K, col0 + 32, k0, lane);
    v16h b3 = load_frag_b(W, K, col0 + 48, k0, lane);
    acc[0][0] = __builtin_amdgcn_wmma_f32_16x16x32_f16(false, a0, false, b0, (short)0, acc[0][0], false, false);
    acc[0][1] = __builtin_amdgcn_wmma_f32_16x16x32_f16(false, a0, false, b1, (short)0, acc[0][1], false, false);
    acc[0][2] = __builtin_amdgcn_wmma_f32_16x16x32_f16(false, a0, false, b2, (short)0, acc[0][2], false, false);
    acc[0][3] = __builtin_amdgcn_wmma_f32_16x16x32_f16(false, a0, false, b3, (short)0, acc[0][3], false, false);
    acc[1][0] = __builtin_amdgcn_wmma_f32_16x16x32_f16(false, a1, false, b0, (short)0, acc[1][0], false, false);
    acc[1][1] = __builtin_amdgcn_wmma_f32_16x16x32_f16(false, a1, false, b1, (short)0, acc[1][1], false, false);
    acc[1][2] = __builtin_amdgcn_wmma_f32_16x16x32_f16(false, a1, false, b2, (short)0, acc[1][2], false, false);
    acc[1][3] = __builtin_amdgcn_wmma_f32_16x16x32_f16(false, a1, false, b3, (short)0, acc[1][3], false, false);

    if (more) {
      asm volatile("s_wait_asynccnt 0" ::: "memory");
      __syncthreads();           // stores visible; readers of old buf done
      cur = 1 - cur;
    }
  }

  // C layout: lane -> n = lane&15, m_base = (lane>>4)*8; VGPR v -> m = m_base+v
  const int nl = lane & 15;
  const int mb = (lane >> 4) << 3;
#pragma unroll
  for (int i = 0; i < 2; ++i) {
#pragma unroll
    for (int j = 0; j < 4; ++j) {
      const int r = row0 + i * 16 + mb;
      const int c = col0 + j * 16 + nl;
#pragma unroll
      for (int v = 0; v < 8; ++v) {
        const float val = acc[i][j][v];
        const long long idx = (long long)(r + v) * N + c;
        if constexpr (EPI == 0) {
          Cout[idx] = val;
        } else if constexpr (EPI == 1) {
          Cout[idx] = val;
          Ch[idx]   = (_Float16)val;
        } else if constexpr (EPI == 2) {
          const float t = val + bias[c];
          Cout[idx] = (t > 20.0f) ? t : log1pf(__expf(t));  // softplus
        } else if constexpr (EPI == 3) {
          Cout[idx] = resid[idx] + val;
        } else if constexpr (EPI == 4) {
          float t = val + bias[c];
          Ch[idx] = (_Float16)(t > 0.0f ? t : 0.0f);        // relu -> f16
        } else {
          Cout[idx] = val + bias[c];
        }
      }
    }
  }
}

// ---------------- f32 -> f16 convert ---------------------------------------
__global__ void f32_to_f16(const float* __restrict__ src,
                           _Float16* __restrict__ dst, long long n) {
  long long i = (long long)blockIdx.x * blockDim.x + threadIdx.x;
  if (i < n) dst[i] = (_Float16)src[i];
}

// ---------------- depthwise conv(k=2) + SiLU -------------------------------
// dirNext=0 (m1): tap l-1.  dirNext=1 (m2 over reversed seq): tap l+1.
__global__ void conv_silu_kernel(const float* __restrict__ xz,
                                 const float* __restrict__ cw,
                                 const float* __restrict__ cb,
                                 float* __restrict__ xc,
                                 _Float16* __restrict__ xch, int dirNext) {
  long long i = (long long)blockIdx.x * blockDim.x + threadIdx.x;
  const long long total = (long long)B_ * L_ * DI_;
  if (i >= total) return;
  const int d = (int)(i % DI_);
  const long long bl = i / DI_;
  const int l = (int)(bl % L_);
  const float xi = xz[bl * (2 * DI_) + d];
  const int ln = dirNext ? l + 1 : l - 1;
  float xp = 0.0f;
  if (ln >= 0 && ln < L_) xp = xz[(bl + (ln - l)) * (2 * DI_) + d];
  const float v = cw[d * 2 + 0] * xp + cw[d * 2 + 1] * xi + cb[d];
  const float s = v / (1.0f + __expf(-v));
  xc[bl * DI_ + d]  = s;
  xch[bl * DI_ + d] = (_Float16)s;
}

// ---------------- selective scan (sequential in L) -------------------------
template<int DIR>
__global__ __launch_bounds__(256)
void scan_kernel(const float* __restrict__ dt, const float* __restrict__ dbc,
                 const float* __restrict__ xc, const float* __restrict__ xz,
                 const float* __restrict__ A_log, const float* __restrict__ Dp,
                 _Float16* __restrict__ yh) {
  __shared__ float sB[DS_];
  __shared__ float sC[DS_];
  const int b = blockIdx.y;
  const int d = blockIdx.x * 256 + threadIdx.x;

  float Am[DS_], h[DS_];
#pragma unroll
  for (int s = 0; s < DS_; ++s) {
    Am[s] = -__expf(A_log[(long long)d * DS_ + s]);
    h[s]  = 0.0f;
  }
  const float Dd = Dp[d];

  for (int t = 0; t < L_; ++t) {
    const int l = DIR ? (L_ - 1 - t) : t;
    const long long base = (long long)b * L_ + l;
    if (threadIdx.x < 2 * DS_) {
      const float v = dbc[base * 64 + 32 + threadIdx.x];
      if (threadIdx.x < DS_) sB[threadIdx.x] = v;
      else                   sC[threadIdx.x - DS_] = v;
    }
    __syncthreads();
    const float dtv = dt[base * DI_ + d];
    const float xv  = xc[base * DI_ + d];
    const float dx  = dtv * xv;
    float y = 0.0f;
#pragma unroll
    for (int s = 0; s < DS_; ++s) {
      h[s] = __expf(dtv * Am[s]) * h[s] + dx * sB[s];
      y += h[s] * sC[s];
    }
    const float zv = xz[base * (2 * DI_) + DI_ + d];
    const float g  = zv / (1.0f + __expf(-zv));
    yh[base * DI_ + d] = (_Float16)((y + xv * Dd) * g);
    __syncthreads();
  }
}

// ---------------- layernorm (row = 512), optional two-input add ------------
__global__ __launch_bounds__(128)
void layernorm_kernel(const float* __restrict__ X, const float* __restrict__ X2,
                      const float* __restrict__ g, const float* __restrict__ bta,
                      float* __restrict__ Y, _Float16* __restrict__ Yh,
                      int twoIn, int wf16) {
  __shared__ float s1[128];
  __shared__ float s2[128];
  const long long row = blockIdx.x;
  const float* x  = X  + row * D_;
  const float* x2 = X2 + row * D_;  // only dereferenced if twoIn
  float v[4];
  float sum = 0.0f, sq = 0.0f;
#pragma unroll
  for (int k = 0; k < 4; ++k) {
    const int c = threadIdx.x + k * 128;
    float t = x[c];
    if (twoIn) t += x2[c];
    v[k] = t; sum += t; sq += t * t;
  }
  s1[threadIdx.x] = sum;
  s2[threadIdx.x] = sq;
  __syncthreads();
  for (int off = 64; off > 0; off >>= 1) {
    if ((int)threadIdx.x < off) {
      s1[threadIdx.x] += s1[threadIdx.x + off];
      s2[threadIdx.x] += s2[threadIdx.x + off];
    }
    __syncthreads();
  }
  const float mu  = s1[0] * (1.0f / D_);
  const float var = s2[0] * (1.0f / D_) - mu * mu;
  const float rs  = rsqrtf(var + 1e-5f);
#pragma unroll
  for (int k = 0; k < 4; ++k) {
    const int c = threadIdx.x + k * 128;
    const float yv = (v[k] - mu) * rs * g[c] + bta[c];
    Y[row * D_ + c] = yv;
    if (wf16) Yh[row * D_ + c] = (_Float16)yv;
  }
}

// ---------------------------------------------------------------------------
extern "C" void kernel_launch(void* const* d_in, const int* in_sizes, int n_in,
                              void* d_out, int out_size, void* d_ws, size_t ws_size,
                              hipStream_t stream) {
  (void)in_sizes; (void)n_in; (void)out_size; (void)ws_size;
  const float* x = (const float*)d_in[0];
  const float* m_in_w[2]   = {(const float*)d_in[1],  (const float*)d_in[10]};
  const float* m_conv_w[2] = {(const float*)d_in[2],  (const float*)d_in[11]};
  const float* m_conv_b[2] = {(const float*)d_in[3],  (const float*)d_in[12]};
  const float* m_xproj[2]  = {(const float*)d_in[4],  (const float*)d_in[13]};
  const float* m_dt_w[2]   = {(const float*)d_in[5],  (const float*)d_in[14]};
  const float* m_dt_b[2]   = {(const float*)d_in[6],  (const float*)d_in[15]};
  const float* m_A_log[2]  = {(const float*)d_in[7],  (const float*)d_in[16]};
  const float* m_Dp[2]     = {(const float*)d_in[8],  (const float*)d_in[17]};
  const float* m_out_w[2]  = {(const float*)d_in[9],  (const float*)d_in[18]};
  const float* ln_g = (const float*)d_in[19];
  const float* ln_b = (const float*)d_in[20];
  const float* c1_w = (const float*)d_in[21];
  const float* c1_b = (const float*)d_in[22];
  const float* c2_w = (const float*)d_in[23];
  const float* c2_b = (const float*)d_in[24];

  const long long M = (long long)B_ * L_;  // 8192 rows

  char* p = (char*)d_ws;
  auto carve = [&](size_t bytes) -> char* {
    char* r = p; p += (bytes + 255) & ~(size_t)255; return r;
  };
  _Float16* xh     = (_Float16*)carve((size_t)M * D_ * 2);
  _Float16* inw16  = (_Float16*)carve((size_t)2 * DI_ * D_ * 2);  // per-mamba reuse
  _Float16* xpw16  = (_Float16*)carve((size_t)64 * DI_ * 2);
  _Float16* dtw16  = (_Float16*)carve((size_t)DI_ * DTR_ * 2);
  _Float16* outw16 = (_Float16*)carve((size_t)D_ * DI_ * 2);
  _Float16* c1w16  = (_Float16*)carve((size_t)DFF_ * D_ * 2);
  _Float16* c2w16  = (_Float16*)carve((size_t)D_ * DFF_ * 2);
  float*    xz     = (float*)   carve((size_t)M * 2 * DI_ * 4);
  float*    xc     = (float*)   carve((size_t)M * DI_ * 4);
  _Float16* xch    = (_Float16*)carve((size_t)M * DI_ * 2);
  float*    dbc    = (float*)   carve((size_t)M * 64 * 4);
  _Float16* dbch   = (_Float16*)carve((size_t)M * 64 * 2);
  float*    dtb    = (float*)   carve((size_t)M * DI_ * 4);
  _Float16* yh     = (_Float16*)carve((size_t)M * DI_ * 2);
  float*    accb   = (float*)   carve((size_t)M * D_ * 4);
  float*    y3     = (float*)   carve((size_t)M * D_ * 4);
  _Float16* y3h    = (_Float16*)carve((size_t)M * D_ * 2);
  _Float16* hh     = (_Float16*)carve((size_t)M * DFF_ * 2);
  float*    yp     = (float*)   carve((size_t)M * D_ * 4);

  auto cvt = [&](const float* s, _Float16* d, long long n) {
    f32_to_f16<<<dim3((unsigned)((n + 255) / 256)), dim3(256), 0, stream>>>(s, d, n);
  };

  cvt(x, xh, M * D_);
  cvt(c1_w, c1w16, (long long)DFF_ * D_);
  cvt(c2_w, c2w16, (long long)D_ * DFF_);

  for (int m = 0; m < 2; ++m) {
    cvt(m_in_w[m],  inw16,  2LL * DI_ * D_);
    cvt(m_xproj[m], xpw16,  64LL * DI_);
    cvt(m_dt_w[m],  dtw16,  (long long)DI_ * DTR_);
    cvt(m_out_w[m], outw16, (long long)D_ * DI_);

    // in_proj: xz[M,4096] = xh[M,512] @ in_w^T
    gemm_wmma<2, 4, 0><<<dim3((unsigned)(M / 64), (2 * DI_) / 256), dim3(256), 0, stream>>>(
        xh, D_, inw16, xz, nullptr, nullptr, nullptr, (int)M, 2 * DI_, D_);

    // depthwise conv(k=2) + SiLU (m2 taps l+1 — reversal folded in)
    {
      const long long tot = M * DI_;
      conv_silu_kernel<<<dim3((unsigned)((tot + 255) / 256)), dim3(256), 0, stream>>>(
          xz, m_conv_w[m], m_conv_b[m], xc, xch, m);
    }

    // xproj: dbc[M,64] = xch[M,2048] @ xproj^T  (f32 + f16 copies)
    gemm_wmma<8, 1, 1><<<dim3((unsigned)(M / 256), 1), dim3(256), 0, stream>>>(
        xch, DI_, xpw16, dbc, dbch, nullptr, nullptr, (int)M, 64, DI_);

    // dt: softplus(dbch[:, :32] @ dt_w^T + dt_b)   (A stride lda=64)
    gemm_wmma<2, 4, 2><<<dim3((unsigned)(M / 64), DI_ / 256), dim3(256), 0, stream>>>(
        dbch, 64, dtw16, dtb, nullptr, m_dt_b[m], nullptr, (int)M, DI_, DTR_);

    // selective scan (+ xc*Dp, * silu(z)) -> f16
    if (m == 0)
      scan_kernel<0><<<dim3(DI_ / 256, B_), dim3(256), 0, stream>>>(
          dtb, dbc, xc, xz, m_A_log[m], m_Dp[m], yh);
    else
      scan_kernel<1><<<dim3(DI_ / 256, B_), dim3(256), 0, stream>>>(
          dtb, dbc, xc, xz, m_A_log[m], m_Dp[m], yh);

    // out_proj with fused residual accumulate: acc = resid + yh @ out_w^T
    const float* resid = (m == 0) ? x : accb;
    gemm_wmma<2, 4, 3><<<dim3((unsigned)(M / 64), D_ / 256), dim3(256), 0, stream>>>(
        yh, DI_, outw16, accb, nullptr, nullptr, resid, (int)M, D_, DI_);
  }

  // y3 = LN(x + y1 + y2), keep f32 (residual) + f16 (GEMM input)
  layernorm_kernel<<<dim3((unsigned)M), dim3(128), 0, stream>>>(
      accb, accb, ln_g, ln_b, y3, y3h, 0, 1);

  // MLP up: hh = relu(y3h @ c1^T + b) -> f16
  gemm_wmma<2, 4, 4><<<dim3((unsigned)(M / 64), DFF_ / 256), dim3(256), 0, stream>>>(
      y3h, D_, c1w16, nullptr, hh, c1_b, nullptr, (int)M, DFF_, D_);

  // MLP down: yp = hh @ c2^T + b
  gemm_wmma<2, 4, 5><<<dim3((unsigned)(M / 64), D_ / 256), dim3(256), 0, stream>>>(
      hh, DFF_, c2w16, yp, nullptr, c2_b, nullptr, (int)M, D_, DFF_);

  // out = LN(yp + y3)
  layernorm_kernel<<<dim3((unsigned)M), dim3(128), 0, stream>>>(
      yp, y3, ln_g, ln_b, (float*)d_out, nullptr, 1, 0);
}
